// GNN_Model_43971875176730
// MI455X (gfx1250) — matrix operand order repaired and checked by reading, not compiled, not run on hardware
//
#include <hip/hip_runtime.h>
#include <hip/hip_bf16.h>

// ---------------- constants ----------------
#define D_DIM 128
#define TD3   384   // 3*D
#define RD    256   // readout width
#define T_ITERS 8

typedef __attribute__((ext_vector_type(16))) __bf16 v16bf;
typedef __attribute__((ext_vector_type(8)))  float  v8f;

union BFrag16 { v16bf v; unsigned short u[16]; };

__device__ __forceinline__ unsigned short f2bf(float f) {
  union { float f; unsigned u; } v; v.f = f;
  unsigned u = v.u;
  u += 0x7FFFu + ((u >> 16) & 1u);   // round-to-nearest-even
  return (unsigned short)(u >> 16);
}
__device__ __forceinline__ float sigmoidf(float x) { return 1.0f / (1.0f + expf(-x)); }

// ---------------- small utility kernels ----------------
__global__ __launch_bounds__(256) void f32_to_bf16_kernel(
    const float* __restrict__ src, unsigned short* __restrict__ dst, int n) {
  int i = blockIdx.x * 256 + threadIdx.x;
  if (i < n) dst[i] = f2bf(src[i]);
}

// transpose-convert: src [Kd][Nd] f32 -> dst [Nd][Kd] bf16
__global__ __launch_bounds__(256) void transpose_f32_to_bf16_kernel(
    const float* __restrict__ src, unsigned short* __restrict__ dst,
    int Kd, int Nd) {
  int i = blockIdx.x * 256 + threadIdx.x;
  if (i < Kd * Nd) {
    int k = i / Nd, n = i - k * Nd;
    dst[(size_t)n * Kd + k] = f2bf(src[i]);
  }
}

__global__ __launch_bounds__(256) void zero_f32_kernel(float* __restrict__ p, int n) {
  int i = blockIdx.x * 256 + threadIdx.x;
  if (i < n) p[i] = 0.0f;
}

// ---------------- state init ----------------
__global__ __launch_bounds__(D_DIM) void init_path_kernel(
    const float* __restrict__ traffic, const float* __restrict__ packets,
    float* __restrict__ F, unsigned short* __restrict__ Bst) {
  int r = blockIdx.x, d = threadIdx.x;
  float v = (d == 0) ? traffic[r] : (d == 1) ? packets[r] : 0.0f;
  size_t idx = (size_t)r * D_DIM + d;
  F[idx] = v; Bst[idx] = f2bf(v);
}

__global__ __launch_bounds__(D_DIM) void init_link_kernel(
    const float* __restrict__ cap, const int* __restrict__ policy,
    float* __restrict__ F, unsigned short* __restrict__ Bst) {
  int r = blockIdx.x, d = threadIdx.x;
  float v = 0.0f;
  if (d == 0) v = cap[r];
  else if (d <= 3) v = (policy[r] == (d - 1)) ? 1.0f : 0.0f;
  size_t idx = (size_t)r * D_DIM + d;
  F[idx] = v; Bst[idx] = f2bf(v);
}

__global__ __launch_bounds__(D_DIM) void init_queue_kernel(
    const float* __restrict__ w, const int* __restrict__ prio,
    float* __restrict__ F, unsigned short* __restrict__ Bst) {
  int r = blockIdx.x, d = threadIdx.x;
  float v = 0.0f;
  if (d < 3) v = (prio[r] == d) ? 1.0f : 0.0f;
  else if (d == 3) v = w[r];
  size_t idx = (size_t)r * D_DIM + d;
  F[idx] = v; Bst[idx] = f2bf(v);
}

// ---------------- path input gather: X[p] = concat(queue[q2p[e]], link[l2p[e]]) ----------------
__global__ __launch_bounds__(D_DIM) void gather_x_kernel(
    const float* __restrict__ qF, const float* __restrict__ lF,
    const int* __restrict__ q2p, const int* __restrict__ l2p,
    unsigned short* __restrict__ X, unsigned char* __restrict__ mask,
    int t, int L) {
  __shared__ int flag;
  int p = blockIdx.x, d = threadIdx.x;
  if (d == 0) flag = 0;
  __syncthreads();
  int e = p * L + t;
  int qi = q2p[e], li = l2p[e];
  float qv = qF[(size_t)qi * D_DIM + d];
  float lv = lF[(size_t)li * D_DIM + d];
  if (qv != 0.0f || lv != 0.0f) flag = 1;   // benign race
  X[(size_t)p * (2 * D_DIM) + d]         = f2bf(qv);
  X[(size_t)p * (2 * D_DIM) + D_DIM + d] = f2bf(lv);
  __syncthreads();
  if (d == 0) mask[p] = (unsigned char)flag;
}

// ---------------- WMMA bf16 GEMM: C[M,N] = A[M,K] @ B[K,N] (+bias, relu) ----------------
// A [M,K] row-major bf16, rows indexed via (row*arow_stride + arow_offset).
// Bt is B PRE-TRANSPOSED: [N][K] row-major bf16. N multiple of 128, K multiple of 128.
#define BM 64
#define BN 128
#define KC 128            // K-slab staged in LDS per stage
#define LDK (KC + 8)

template <int K>
__global__ __launch_bounds__(256) void gemm_bf16_kernel(
    const unsigned short* __restrict__ A, int arow_stride, int arow_offset,
    const unsigned short* __restrict__ Bt, const float* __restrict__ bias,
    float* __restrict__ Cf, unsigned short* __restrict__ Cb,
    int M, int N, int relu) {
  __shared__ unsigned short a_s[BM][LDK];   // 64 x 136 -> 17 KB
  __shared__ unsigned short b_s[BN][LDK];   // 128 x 136 -> 34 KB

  const int tid  = threadIdx.x;
  const int lane = tid & 31;
  const int wave = tid >> 5;
  const int bm = blockIdx.y * BM;
  const int bn = blockIdx.x * BN;
  const int mt  = wave & 3;          // M-tile 0..3
  const int ntg = (wave >> 2) * 4;   // N-tile group base: 0 or 4
  const int l15   = lane & 15;
  const int khalf = lane >> 4;

  v8f acc[4] = {{}, {}, {}, {}};

  // A staging geometry: thread -> (row = tid/4, col groups of 8 stride 32)
  const int ar  = tid >> 2;
  const int ac0 = (tid & 3) * 8;
  const bool a_ok = (bm + ar) < M;
  const size_t a_memrow = (size_t)(bm + ar) * (size_t)arow_stride + (size_t)arow_offset;
  const unsigned short* a_src = A + a_memrow * (size_t)K;
  // B staging geometry: thread -> (n = tid/2, col groups of 8 stride 16)
  const int br  = tid >> 1;
  const int bc0 = (tid & 1) * 8;
  const unsigned short* b_src = Bt + (size_t)(bn + br) * (size_t)K;

#pragma unroll
  for (int ks = 0; ks < K; ks += KC) {
    // ---- stage A slab (BM x KC), 16B vector ops both sides ----
#pragma unroll
    for (int c = ac0; c < KC; c += 32) {
      uint4 v = make_uint4(0u, 0u, 0u, 0u);
      if (a_ok) v = *reinterpret_cast<const uint4*>(a_src + ks + c);
      *reinterpret_cast<uint4*>(&a_s[ar][c]) = v;
    }
    // ---- stage Bt slab (BN x KC) ----
#pragma unroll
    for (int c = bc0; c < KC; c += 16) {
      *reinterpret_cast<uint4*>(&b_s[br][c]) =
          *reinterpret_cast<const uint4*>(b_src + ks + c);
    }
    __syncthreads();

    // ---- barrier-free compute burst over this K slab ----
    const unsigned short* arow = &a_s[mt * 16 + l15][0];
#pragma unroll
    for (int kk = 0; kk < KC; kk += 32) {
      // Load A fragment and ALL FOUR B fragments into distinct registers first,
      // so the scheduler can overlap ds_loads across the 4 back-to-back WMMAs.
      BFrag16 a;
#pragma unroll
      for (int i = 0; i < 8; ++i) {
        a.u[i]     = arow[kk + khalf * 16 + i];       // K = kk + khalf*16 + i
        a.u[8 + i] = arow[kk + khalf * 16 + 8 + i];   // K = kk + khalf*16 + 8 + i
      }
      BFrag16 b[4];
#pragma unroll
      for (int j = 0; j < 4; ++j) {
        const unsigned short* brow = &b_s[(ntg + j) * 16 + l15][kk + khalf * 16];
#pragma unroll
        for (int i = 0; i < 16; ++i) b[j].u[i] = brow[i];
      }
      acc[0] = __builtin_amdgcn_wmma_f32_16x16x32_bf16(false, a.v, false, b[0].v,
                                                       (short)0, acc[0], false, false);
      acc[1] = __builtin_amdgcn_wmma_f32_16x16x32_bf16(false, a.v, false, b[1].v,
                                                       (short)0, acc[1], false, false);
      acc[2] = __builtin_amdgcn_wmma_f32_16x16x32_bf16(false, a.v, false, b[2].v,
                                                       (short)0, acc[2], false, false);
      acc[3] = __builtin_amdgcn_wmma_f32_16x16x32_bf16(false, a.v, false, b[3].v,
                                                       (short)0, acc[3], false, false);
    }
    __syncthreads();
  }

  // ---- epilogue (C layout: VGPR r -> M = khalf*8 + r, N = l15) ----
#pragma unroll
  for (int r = 0; r < 8; ++r) {
    int row = bm + mt * 16 + khalf * 8 + r;
    if (row < M) {
#pragma unroll
      for (int j = 0; j < 4; ++j) {
        int col = bn + (ntg + j) * 16 + l15;
        float v = acc[j][r] + (bias ? bias[col] : 0.0f);
        if (relu) v = v > 0.0f ? v : 0.0f;
        Cf[(size_t)row * N + col] = v;
        if (Cb) Cb[(size_t)row * N + col] = f2bf(v);
      }
    }
  }
}

// ---------------- GRU pointwise ----------------
// mask modes: mask!=null -> use mask array; else xsrc!=null -> any(x!=0); else always update.
__global__ __launch_bounds__(D_DIM) void gru_pointwise_kernel(
    const float* __restrict__ gi, const float* __restrict__ gh,
    float* __restrict__ hF, unsigned short* __restrict__ hB,
    const unsigned char* __restrict__ mask,
    const float* __restrict__ xsrc, int xstride, int xoff) {
  __shared__ int flag;
  int row = blockIdx.x, d = threadIdx.x;
  if (d == 0) flag = (mask || xsrc) ? 0 : 1;
  __syncthreads();
  if (mask) {
    if (d == 0) flag = (int)mask[row];
  } else if (xsrc) {
    float xv = xsrc[((size_t)row * xstride + xoff) * D_DIM + d];
    if (xv != 0.0f) flag = 1;   // benign race
  }
  __syncthreads();

  size_t b3 = (size_t)row * TD3;
  float iz = gi[b3 + d], ir = gi[b3 + D_DIM + d], ih = gi[b3 + 2 * D_DIM + d];
  float hz = gh[b3 + d], hr = gh[b3 + D_DIM + d], hh = gh[b3 + 2 * D_DIM + d];
  size_t hidx = (size_t)row * D_DIM + d;
  float h = hF[hidx];
  float z = sigmoidf(iz + hz);
  float r = sigmoidf(ir + hr);
  float cand = tanhf(ih + r * hh);
  float hn = z * h + (1.0f - z) * cand;
  float out = flag ? hn : h;
  hF[hidx] = out;
  hB[hidx] = f2bf(out);
}

// ---------------- segment-sum scatter ----------------
__global__ __launch_bounds__(D_DIM) void scatter_psum_kernel(
    const float* __restrict__ pathF, const int* __restrict__ pidx,
    const int* __restrict__ qidx, float* __restrict__ psum) {
  int e = blockIdx.x, d = threadIdx.x;
  int p = pidx[e], q = qidx[e];
  atomicAdd(&psum[(size_t)q * D_DIM + d], pathF[(size_t)p * D_DIM + d]);
}

// ---------------- final readout: out = H2 @ Wr3 + br3 (N=1) ----------------
__global__ __launch_bounds__(256) void final_readout_kernel(
    const float* __restrict__ H2, const float* __restrict__ Wr3,
    const float* __restrict__ br3, float* __restrict__ out, int M) {
  int p = blockIdx.x * 256 + threadIdx.x;
  if (p >= M) return;
  float s = 0.0f;
#pragma unroll 8
  for (int j = 0; j < RD; ++j) s += H2[(size_t)p * RD + j] * Wr3[j];
  out[p] = s + br3[0];
}

// ---------------- host orchestration ----------------
extern "C" void kernel_launch(void* const* d_in, const int* in_sizes, int n_in,
                              void* d_out, int out_size, void* d_ws, size_t ws_size,
                              hipStream_t stream) {
  const float* traffic  = (const float*)d_in[0];
  const float* packets  = (const float*)d_in[1];
  const float* capacity = (const float*)d_in[2];
  const float* weight   = (const float*)d_in[3];
  const int*   policy   = (const int*)d_in[4];
  const int*   priority = (const int*)d_in[5];
  const int*   link_to_path    = (const int*)d_in[9];
  const int*   queue_to_path   = (const int*)d_in[10];
  const int*   path_to_queue   = (const int*)d_in[11];
  const int*   sequence_queues = (const int*)d_in[12];
  const float* Wp = (const float*)d_in[19];
  const float* Up = (const float*)d_in[20];
  const float* bp = (const float*)d_in[21];
  const float* Wl = (const float*)d_in[22];
  const float* Ul = (const float*)d_in[23];
  const float* bl = (const float*)d_in[24];
  const float* Wq = (const float*)d_in[25];
  const float* Uq = (const float*)d_in[26];
  const float* bq = (const float*)d_in[27];
  const float* Wr1 = (const float*)d_in[28];
  const float* br1 = (const float*)d_in[29];
  const float* Wr2 = (const float*)d_in[30];
  const float* br2 = (const float*)d_in[31];
  const float* Wr3 = (const float*)d_in[32];
  const float* br3 = (const float*)d_in[33];

  const int NP = in_sizes[0];
  const int NL = in_sizes[2];
  const int NQ = in_sizes[3];
  const int E  = in_sizes[6];
  const int L  = E / NP;       // 8
  const int QLn = NQ / NL;     // 3

  // ---- workspace carve ----
  char* base = (char*)d_ws;
  size_t off = 0;
  auto carve = [&](size_t bytes) -> void* {
    void* p = base + off;
    off += (bytes + 255) & ~(size_t)255;
    return p;
  };
  float*          pathF  = (float*)carve((size_t)NP * D_DIM * 4);
  unsigned short* pathB  = (unsigned short*)carve((size_t)NP * D_DIM * 2);
  float*          linkF  = (float*)carve((size_t)NL * D_DIM * 4);
  unsigned short* linkB  = (unsigned short*)carve((size_t)NL * D_DIM * 2);
  float*          queueF = (float*)carve((size_t)NQ * D_DIM * 4);
  unsigned short* queueB = (unsigned short*)carve((size_t)NQ * D_DIM * 2);
  float*          psumF  = (float*)carve((size_t)NQ * D_DIM * 4);
  unsigned short* psumB  = (unsigned short*)carve((size_t)NQ * D_DIM * 2);
  unsigned short* Xb     = (unsigned short*)carve((size_t)NP * 2 * D_DIM * 2); // also H1 bf16
  unsigned char*  maskP  = (unsigned char*)carve((size_t)NP);
  float*          GI     = (float*)carve((size_t)NP * TD3 * 4);  // also H1 f32
  float*          GH     = (float*)carve((size_t)NP * TD3 * 4);  // also H2 f32
  // pre-transposed bf16 weights: [N][K]
  unsigned short* WpT  = (unsigned short*)carve((size_t)2 * D_DIM * TD3 * 2);
  unsigned short* UpT  = (unsigned short*)carve((size_t)D_DIM * TD3 * 2);
  unsigned short* WlT  = (unsigned short*)carve((size_t)D_DIM * TD3 * 2);
  unsigned short* UlT  = (unsigned short*)carve((size_t)D_DIM * TD3 * 2);
  unsigned short* WqT  = (unsigned short*)carve((size_t)D_DIM * TD3 * 2);
  unsigned short* UqT  = (unsigned short*)carve((size_t)D_DIM * TD3 * 2);
  unsigned short* Wr1T = (unsigned short*)carve((size_t)D_DIM * RD * 2);
  unsigned short* Wr2T = (unsigned short*)carve((size_t)RD * RD * 2);
  (void)ws_size; (void)n_in;

  auto cvtT = [&](const float* s, unsigned short* d, int Kd, int Nd) {
    int n = Kd * Nd;
    transpose_f32_to_bf16_kernel<<<(n + 255) / 256, 256, 0, stream>>>(s, d, Kd, Nd);
  };

  // ---- one-time conversions (transpose weights to [N][K]) + state init ----
  cvtT(Wp, WpT, 2 * D_DIM, TD3);
  cvtT(Up, UpT, D_DIM, TD3);
  cvtT(Wl, WlT, D_DIM, TD3);
  cvtT(Ul, UlT, D_DIM, TD3);
  cvtT(Wq, WqT, D_DIM, TD3);
  cvtT(Uq, UqT, D_DIM, TD3);
  cvtT(Wr1, Wr1T, D_DIM, RD);
  cvtT(Wr2, Wr2T, RD, RD);
  init_path_kernel<<<NP, D_DIM, 0, stream>>>(traffic, packets, pathF, pathB);
  init_link_kernel<<<NL, D_DIM, 0, stream>>>(capacity, policy, linkF, linkB);
  init_queue_kernel<<<NQ, D_DIM, 0, stream>>>(weight, priority, queueF, queueB);

  dim3 gP(TD3 / BN, (NP + BM - 1) / BM);
  dim3 gQ(TD3 / BN, (NQ + BM - 1) / BM);
  dim3 gL(TD3 / BN, (NL + BM - 1) / BM);

  for (int it = 0; it < T_ITERS; ++it) {
    // ---- path masked GRU over L timesteps ----
    for (int t = 0; t < L; ++t) {
      gather_x_kernel<<<NP, D_DIM, 0, stream>>>(queueF, linkF, queue_to_path,
                                                link_to_path, Xb, maskP, t, L);
      gemm_bf16_kernel<2 * D_DIM><<<gP, 256, 0, stream>>>(
          Xb, 1, 0, WpT, bp, GI, nullptr, NP, TD3, 0);
      gemm_bf16_kernel<D_DIM><<<gP, 256, 0, stream>>>(
          pathB, 1, 0, UpT, bp + TD3, GH, nullptr, NP, TD3, 0);
      gru_pointwise_kernel<<<NP, D_DIM, 0, stream>>>(
          GI, GH, pathF, pathB, maskP, nullptr, 0, 0);
    }
    // ---- queue GRU (single step on segment-summed path states) ----
    zero_f32_kernel<<<(NQ * D_DIM + 255) / 256, 256, 0, stream>>>(psumF, NQ * D_DIM);
    scatter_psum_kernel<<<E, D_DIM, 0, stream>>>(pathF, path_to_queue,
                                                 sequence_queues, psumF);
    f32_to_bf16_kernel<<<(NQ * D_DIM + 255) / 256, 256, 0, stream>>>(
        psumF, psumB, NQ * D_DIM);
    gemm_bf16_kernel<D_DIM><<<gQ, 256, 0, stream>>>(
        psumB, 1, 0, WqT, bq, GI, nullptr, NQ, TD3, 0);
    gemm_bf16_kernel<D_DIM><<<gQ, 256, 0, stream>>>(
        queueB, 1, 0, UqT, bq + TD3, GH, nullptr, NQ, TD3, 0);
    gru_pointwise_kernel<<<NQ, D_DIM, 0, stream>>>(
        GI, GH, queueF, queueB, nullptr, nullptr, 0, 0);
    // ---- link masked GRU over QL timesteps (inputs = strided queue rows) ----
    for (int t = 0; t < QLn; ++t) {
      gemm_bf16_kernel<D_DIM><<<gL, 256, 0, stream>>>(
          queueB, QLn, t, WlT, bl, GI, nullptr, NL, TD3, 0);
      gemm_bf16_kernel<D_DIM><<<gL, 256, 0, stream>>>(
          linkB, 1, 0, UlT, bl + TD3, GH, nullptr, NL, TD3, 0);
      gru_pointwise_kernel<<<NL, D_DIM, 0, stream>>>(
          GI, GH, linkF, linkB, nullptr, queueF, QLn, t);
    }
  }

  // ---- readout MLP ----
  dim3 gR(RD / BN, (NP + BM - 1) / BM);
  // H1 = relu(path @ Wr1 + br1): f32 in GI, bf16 mirror in Xb
  gemm_bf16_kernel<D_DIM><<<gR, 256, 0, stream>>>(
      pathB, 1, 0, Wr1T, br1, GI, Xb, NP, RD, 1);
  // H2 = relu(H1 @ Wr2 + br2): f32 in GH
  gemm_bf16_kernel<RD><<<gR, 256, 0, stream>>>(
      Xb, 1, 0, Wr2T, br2, GH, nullptr, NP, RD, 1);
  final_readout_kernel<<<(NP + 255) / 256, 256, 0, stream>>>(
      GH, Wr3, br3, (float*)d_out, NP);
}
